// LinearFLHGSWrapper_1176821039703
// MI455X (gfx1250) — 8-bit matrix operand order repaired and checked by reading, compiled, not run on hardware
//
#include <hip/hip_runtime.h>
#include <stdint.h>

typedef __attribute__((ext_vector_type(8))) int  v8i;
typedef __attribute__((ext_vector_type(4))) unsigned int u32x4;
typedef __attribute__((ext_vector_type(8))) int  i32x8;
typedef __attribute__((ext_vector_type(4))) int  i32x4;

#define KDIM 4096
#define GS   64
#define NGRP (KDIM / GS)
#define BM   128
#define BN   128
#define KK   128
#define NIT  (KDIM / KK)
#define LROW 144
#define SLAB (BM * LROW)
#define BUFSZ (2 * SLAB)
#define SMEM_BYTES (2 * BUFSZ)

__global__ __launch_bounds__(256) void quant_i4_gs64(
    const float* __restrict__ in, int8_t* __restrict__ q,
    float* __restrict__ scales, int rows)
{
    int gwave = (blockIdx.x * blockDim.x + threadIdx.x) >> 5;
    int lane  = threadIdx.x & 31;
    if (gwave >= rows * NGRP) return;
    int row = gwave >> 6;
    int g   = gwave & (NGRP - 1);

    size_t base = (size_t)row * KDIM + (size_t)g * GS + (size_t)lane * 2;
    float2 v = *(const float2*)(in + base);
    float m = fmaxf(fabsf(v.x), fabsf(v.y));
    #pragma unroll
    for (int off = 16; off > 0; off >>= 1)
        m = fmaxf(m, __shfl_xor(m, off, 32));
    float s = fmaxf(m * (1.0f / 7.0f), 1e-8f);
    float qa = fminf(fmaxf(rintf(v.x / s), -8.0f), 7.0f);
    float qb = fminf(fmaxf(rintf(v.y / s), -8.0f), 7.0f);
    char2 c;
    c.x = (signed char)(int)qa;
    c.y = (signed char)(int)qb;
    *(char2*)(q + base) = c;
    if (lane == 0) scales[(size_t)g * rows + row] = s;
}

__device__ __forceinline__ i32x8 tdm_make_g1(
    uint32_t tensor_d0, uint32_t tensor_d1, uint64_t stride0,
    uint32_t tile_d0, uint32_t tile_d1)
{
    i32x8 g1;
    g1[0] = (int)((1u << 20) | (4u << 22) | (3u << 25));
    g1[1] = (int)((tensor_d0 & 0xFFFFu) << 16);
    g1[2] = (int)((tensor_d0 >> 16) | ((tensor_d1 & 0xFFFFu) << 16));
    g1[3] = (int)((tensor_d1 >> 16) | (tile_d0 << 16));
    g1[4] = (int)(tile_d1 & 0xFFFFu);
    g1[5] = (int)(uint32_t)stride0;
    g1[6] = (int)(uint32_t)((stride0 >> 32) & 0xFFFFu);
    g1[7] = 0;
    return g1;
}

__device__ __forceinline__ void tdm_issue(uint32_t lds_addr, uint64_t gaddr,
                                          i32x8 g1)
{
    u32x4 g0;
    g0[0] = 1u;
    g0[1] = lds_addr;
    g0[2] = (uint32_t)gaddr;
    g0[3] = (uint32_t)(gaddr >> 32) | (2u << 30);
    i32x4 z4 = {0, 0, 0, 0};
    i32x8 z8 = {0, 0, 0, 0, 0, 0, 0, 0};
    __builtin_amdgcn_tensor_load_to_lds(g0, g1, z4, z4, z8, 0);
}

__global__ __launch_bounds__(256) void gemm_i4_gs64_wmma(
    const int8_t* __restrict__ xq, const float* __restrict__ sx,
    const int8_t* __restrict__ wq, const float* __restrict__ sw,
    float* __restrict__ out, int T, int O)
{
    extern __shared__ int8_t smem[];

    const int tid   = threadIdx.x;
    const int wid   = tid >> 5;
    const int lane  = tid & 31;
    const int half  = lane >> 4;
    const int nn    = lane & 15;
    const int row0  = blockIdx.y * BM;
    const int col0  = blockIdx.x * BN;
    const int m_off = (wid >> 1) * 32;
    const int n_off = (wid & 1) * 64;

    const uint32_t lds_base = (uint32_t)(uintptr_t)(void*)smem;

    const i32x8 g1A = tdm_make_g1(KDIM, (uint32_t)T, KDIM, KK, BM);
    const i32x8 g1B = tdm_make_g1(KDIM, (uint32_t)O, KDIM, KK, BN);
    uint64_t gA = (uint64_t)(uintptr_t)(xq + (size_t)row0 * KDIM);
    uint64_t gB = (uint64_t)(uintptr_t)(wq + (size_t)col0 * KDIM);

    float acc[2][4][8];
    #pragma unroll
    for (int mt = 0; mt < 2; ++mt)
        #pragma unroll
        for (int nt = 0; nt < 4; ++nt)
            #pragma unroll
            for (int j = 0; j < 8; ++j)
                acc[mt][nt][j] = 0.0f;

    if (wid == 0) {
        tdm_issue(lds_base + 0 * BUFSZ, gA, g1A);
        tdm_issue(lds_base + 0 * BUFSZ + SLAB, gB, g1B);
    }

    for (int kk = 0; kk < NIT; ++kk) {
        const int cur = kk & 1;

        if (wid == 0) {
            if (kk + 1 < NIT) {
                gA += KK;
                gB += KK;
                tdm_issue(lds_base + (1 - cur) * BUFSZ, gA, g1A);
                tdm_issue(lds_base + (1 - cur) * BUFSZ + SLAB, gB, g1B);
                __builtin_amdgcn_s_wait_tensorcnt((short)2);
            } else {
                __builtin_amdgcn_s_wait_tensorcnt((short)0);
            }
        }
        __syncthreads();

        const int8_t* A0 = smem + cur * BUFSZ;
        const int8_t* B0 = smem + cur * BUFSZ + SLAB;

        v8i afrag[2][2];
        #pragma unroll
        for (int g2 = 0; g2 < 2; ++g2)
            #pragma unroll
            for (int mt = 0; mt < 2; ++mt) {
                const int8_t* ab = A0 + (m_off + mt * 16 + nn) * LROW + g2 * 64;
                #pragma unroll
                for (int vv = 0; vv < 4; ++vv) {
                    int2 p = *(const int2*)(ab + (vv >> 1) * 32 + half * 16 + (vv & 1) * 8);
                    afrag[g2][mt][2 * vv]     = p.x;
                    afrag[g2][mt][2 * vv + 1] = p.y;
                }
            }

        float sxv[2][2][8];
        float swv[2][4];
        #pragma unroll
        for (int g2 = 0; g2 < 2; ++g2) {
            const int g = kk * 2 + g2;
            #pragma unroll
            for (int mt = 0; mt < 2; ++mt) {
                const float* sp = sx + (size_t)g * T +
                                  (row0 + m_off + mt * 16 + half * 8);
                float4 lo = *(const float4*)(sp);
                float4 hi = *(const float4*)(sp + 4);
                sxv[g2][mt][0] = lo.x; sxv[g2][mt][1] = lo.y;
                sxv[g2][mt][2] = lo.z; sxv[g2][mt][3] = lo.w;
                sxv[g2][mt][4] = hi.x; sxv[g2][mt][5] = hi.y;
                sxv[g2][mt][6] = hi.z; sxv[g2][mt][7] = hi.w;
            }
            #pragma unroll
            for (int nt = 0; nt < 4; ++nt)
                swv[g2][nt] = sw[(size_t)g * O + (col0 + n_off + nt * 16 + nn)];
        }

        auto fixup = [&](int p, v8i d) {
            const int pg2 = p >> 3, pnt = (p >> 1) & 3, pmt = p & 1;
            float pf[8], df[8];
            #pragma unroll
            for (int j = 0; j < 8; ++j)
                pf[j] = sxv[pg2][pmt][j] * swv[pg2][pnt];
            #pragma unroll
            for (int j = 0; j < 8; ++j)
                df[j] = (float)d[j];
            #pragma unroll
            for (int j = 0; j < 8; ++j)
                acc[pmt][pnt][j] += df[j] * pf[j];
        };

        v8i dbuf[4];
        v8i bcur;
        #pragma unroll
        for (int idx = 0; idx < 16; ++idx) {
            const int g2 = idx >> 3, nt = (idx >> 1) & 3, mt = idx & 1;
            if (mt == 0) {
                const int8_t* bb = B0 + (n_off + nt * 16 + nn) * LROW + g2 * 64;
                int4 q0 = *(const int4*)(bb + half * 16);
                int4 q1 = *(const int4*)(bb + 32 + half * 16);
                bcur[0] = q0.x; bcur[1] = q0.y; bcur[2] = q0.z; bcur[3] = q0.w;
                bcur[4] = q1.x; bcur[5] = q1.y; bcur[6] = q1.z; bcur[7] = q1.w;
            }
            v8i c = {};
            dbuf[idx & 3] = __builtin_amdgcn_wmma_i32_16x16x64_iu8(
                                true, afrag[g2][mt], true, bcur, c, false, false);
            if (idx >= 2)
                fixup(idx - 2, dbuf[(idx - 2) & 3]);
        }
        fixup(14, dbuf[14 & 3]);
        fixup(15, dbuf[15 & 3]);
        __syncthreads();
    }

    #pragma unroll
    for (int mt = 0; mt < 2; ++mt)
        #pragma unroll
        for (int nt = 0; nt < 4; ++nt)
            #pragma unroll
            for (int j = 0; j < 8; ++j)
                out[(size_t)(row0 + m_off + mt * 16 + half * 8 + j) * O +
                    col0 + n_off + nt * 16 + nn] = acc[mt][nt][j];
}

extern "C" void kernel_launch(void* const* d_in, const int* in_sizes, int n_in,
                              void* d_out, int out_size, void* d_ws, size_t ws_size,
                              hipStream_t stream)
{
    const float* x = (const float*)d_in[0];
    const float* w = (const float*)d_in[1];
    float* out = (float*)d_out;

    const int T = in_sizes[0] / KDIM;
    const int O = in_sizes[1] / KDIM;

    uint8_t* ws = (uint8_t*)d_ws;
    int8_t* xq = (int8_t*)ws;
    int8_t* wq = (int8_t*)(ws + (size_t)T * KDIM);
    float*  sx = (float*)(ws + (size_t)T * KDIM + (size_t)O * KDIM);
    float*  sw = sx + (size_t)T * NGRP;

    {
        int waves  = T * NGRP;
        int blocks = (waves * 32 + 255) / 256;
        quant_i4_gs64<<<blocks, 256, 0, stream>>>(x, xq, sx, T);
    }
    {
        int waves  = O * NGRP;
        int blocks = (waves * 32 + 255) / 256;
        quant_i4_gs64<<<blocks, 256, 0, stream>>>(w, wq, sw, O);
    }

    dim3 grid(O / BN, T / BM);
    gemm_i4_gs64_wmma<<<grid, 256, SMEM_BYTES, stream>>>(xq, sx, wq, sw, out, T, O);
}
